// MultiHead_1073741824211
// MI455X (gfx1250) — matrix operand order repaired and checked by reading, not compiled, not run on hardware
//
#include <hip/hip_runtime.h>
#include <hip/hip_bf16.h>

// ---- problem constants ----
#define DM 256   // d_model
#define DH 64    // d_head
#define NH 4     // heads
#define BB 8     // batch
#define NN 2048  // seq len
#define LDP 40   // LDS row stride in halves for P tiles (16B-aligned, no bank conflicts)

typedef __attribute__((ext_vector_type(16))) _Float16 v16h;
typedef __attribute__((ext_vector_type(8)))  _Float16 v8h;
typedef __attribute__((ext_vector_type(8)))  float    v8f;
typedef __attribute__((ext_vector_type(4)))  float    v4f;

__device__ __forceinline__ v8f wmma_f16(v16h a, v16h b, v8f c) {
  return __builtin_amdgcn_wmma_f32_16x16x32_f16(
      false, a, false, b, (short)0, c, false, false);
}

__device__ __forceinline__ v16h pack16(v8h lo, v8h hi) {
  v16h r;
#pragma unroll
  for (int i = 0; i < 8; i++) { r[i] = lo[i]; r[i + 8] = hi[i]; }
  return r;
}

// A fragment (16x32 f16) from row-major f16 [rows x ld].
__device__ __forceinline__ v16h load_a_f16(const _Float16* base, int ld, int kb, int lane) {
  int row = lane & 15, koff = (lane >> 4) << 3;
  const _Float16* p = base + (size_t)row * ld + kb + koff;
  v8h lo = *(const v8h*)(p);
  v8h hi = *(const v8h*)(p + 16);
  return pack16(lo, hi);
}

// A fragment from row-major f32, converted to f16 in-registers.
__device__ __forceinline__ v16h load_a_f32(const float* base, int ld, int kb, int lane) {
  int row = lane & 15, koff = (lane >> 4) << 3;
  const float* p = base + (size_t)row * ld + kb + koff;
  v4f x0 = *(const v4f*)(p);
  v4f x1 = *(const v4f*)(p + 4);
  v4f x2 = *(const v4f*)(p + 16);
  v4f x3 = *(const v4f*)(p + 20);
  v16h r;
#pragma unroll
  for (int i = 0; i < 4; i++) {
    r[i]      = (_Float16)x0[i];
    r[i + 4]  = (_Float16)x1[i];
    r[i + 8]  = (_Float16)x2[i];
    r[i + 12] = (_Float16)x3[i];
  }
  return r;
}

// B fragment (32x16 f16) where element (k, col) lives at base[col*ld + k].
__device__ __forceinline__ v16h load_b_f16(const _Float16* base, int ld, int kb, int lane) {
  int col = lane & 15, k0 = (lane >> 4) << 3;
  const _Float16* p = base + (size_t)col * ld + kb + k0;
  v8h lo = *(const v8h*)(p);
  v8h hi = *(const v8h*)(p + 16);
  return pack16(lo, hi);
}

// Same, from f32 source with conversion.
__device__ __forceinline__ v16h load_b_f32(const float* base, int ld, int kb, int lane) {
  int col = lane & 15, k0 = (lane >> 4) << 3;
  const float* p = base + (size_t)col * ld + kb + k0;
  v16h r;
#pragma unroll
  for (int j = 0; j < 4; j++) {
    v4f x = *(const v4f*)(p + 4 * j + 8 * (j >> 1));
#pragma unroll
    for (int i = 0; i < 4; i++) r[4 * j + i] = (_Float16)x[i];
  }
  return r;
}

// ---- CDNA5 async global->LDS copy (16B per lane), tracked by ASYNCcnt ----
__device__ __forceinline__ unsigned lds_off(const void* p) {
  // addrspace(3) offset lives in the low 32 bits of the generic pointer
  return (unsigned)(unsigned long long)p;
}
__device__ __forceinline__ void async_ld16(unsigned ldsaddr, const void* gaddr) {
  asm volatile("global_load_async_to_lds_b128 %0, %1, off"
               :: "v"(ldsaddr), "v"(gaddr) : "memory");
}
__device__ __forceinline__ void wait_async0() {
#if __has_builtin(__builtin_amdgcn_s_wait_asynccnt)
  __builtin_amdgcn_s_wait_asynccnt(0);
#else
  asm volatile("s_wait_asynccnt 0x0" ::: "memory");
#endif
}

// ---------------------------------------------------------------------------
// Kernel 0: weights [h,256,64] f32 -> transposed [t,h,64,256] f16
// ---------------------------------------------------------------------------
__global__ __launch_bounds__(256) void prep_weights(const float* Wq, const float* Wk,
                                                    const float* Wv, const float* Wr,
                                                    _Float16* WtT) {
  int tid = blockIdx.x * blockDim.x + threadIdx.x;  // 0..262143
  int t = tid >> 16;
  int rem = tid & 65535;
  int h = rem >> 14;
  int rem2 = rem & 16383;
  int k = rem2 >> 6;
  int c = rem2 & 63;
  const float* W = (t == 0) ? Wq : (t == 1) ? Wk : (t == 2) ? Wv : Wr;
  WtT[(((size_t)(t * NH + h) * DH + c) << 8) + k] = (_Float16)W[((size_t)h * DM + k) * DH + c];
}

// ---------------------------------------------------------------------------
// Kernel 1: projections (one wave per 16x16 output tile).
// job 0: q -> qh [b,h,n,64] f16; job 1: k+r -> krh; job 2: v -> vT [b,h,64,n] f16
// ---------------------------------------------------------------------------
__global__ __launch_bounds__(256) void proj_kernel(
    const float* Q, const float* K, const float* V, const float* R,
    const float* bq, const float* bk, const float* bv, const float* br,
    const _Float16* WtT, _Float16* qh, _Float16* krh, _Float16* vT) {
  int w = blockIdx.x * (blockDim.x >> 5) + (threadIdx.x >> 5);
  int lane = threadIdx.x & 31;
  int job = w >> 14;
  int r = w & 16383;
  int ct = r & 3;
  int h  = (r >> 2) & 3;
  int nb = (r >> 4) & 127;
  int b  = (r >> 11) & 7;
  int col = lane & 15, hi = lane >> 4;
  int hc = h * DH + ct * 16 + col;

  v8f c = {};
  if (job == 1) {
    const float* A1 = K + ((size_t)b * NN + nb * 16) * DM;
    const float* A2 = R + ((size_t)b * NN + nb * 16) * DM;
    const _Float16* B1 = WtT + ((size_t)(1 * NH + h) * DH + ct * 16) * DM;
    const _Float16* B2 = WtT + ((size_t)(3 * NH + h) * DH + ct * 16) * DM;
#pragma unroll
    for (int kc = 0; kc < 8; kc++) {
      c = wmma_f16(load_a_f32(A1, DM, kc * 32, lane), load_b_f16(B1, DM, kc * 32, lane), c);
      c = wmma_f16(load_a_f32(A2, DM, kc * 32, lane), load_b_f16(B2, DM, kc * 32, lane), c);
    }
    float bias = bk[hc] + br[hc];
    _Float16* o = krh + ((size_t)(b * NH + h) * NN + nb * 16) * DH + ct * 16;
#pragma unroll
    for (int g = 0; g < 8; g++)
      o[(size_t)(g + hi * 8) * DH + col] = (_Float16)(c[g] + bias);
  } else {
    const float* A = (job == 0 ? Q : V) + ((size_t)b * NN + nb * 16) * DM;
    const _Float16* Bw = WtT + ((size_t)((job == 0 ? 0 : 2) * NH + h) * DH + ct * 16) * DM;
#pragma unroll
    for (int kc = 0; kc < 8; kc++)
      c = wmma_f16(load_a_f32(A, DM, kc * 32, lane), load_b_f16(Bw, DM, kc * 32, lane), c);
    if (job == 0) {
      float bias = bq[hc];
      _Float16* o = qh + ((size_t)(b * NH + h) * NN + nb * 16) * DH + ct * 16;
#pragma unroll
      for (int g = 0; g < 8; g++)
        o[(size_t)(g + hi * 8) * DH + col] = (_Float16)(c[g] + bias);
    } else {
      float bias = bv[hc];
      v8h pk;
#pragma unroll
      for (int g = 0; g < 8; g++) pk[g] = (_Float16)(c[g] + bias);
      _Float16* o = vT + ((size_t)(b * NH + h) * DH + ct * 16 + col) * NN + nb * 16 + hi * 8;
      *(v8h*)o = pk;
    }
  }
}

// ---------------------------------------------------------------------------
// Kernel 2: attention per (b, h, 16-query block); 4 waves/block share (b,h).
// kr / vT tiles are staged into LDS once per block via async copies
// (double-buffered), scores streamed to HBM with non-temporal stores.
// ---------------------------------------------------------------------------
__global__ __launch_bounds__(128) void attn_kernel(const _Float16* qh, const _Float16* krh,
                                                   const _Float16* vT, float* scores,
                                                   _Float16* xbuf) {
  __shared__ _Float16 kr1[2][16 * DH];   // pass-1 staging: 16 keys x 64   (2x2KB)
  __shared__ _Float16 kr2[2][32 * DH];   // pass-2 staging: 32 keys x 64   (2x4KB)
  __shared__ _Float16 vt2[2][DH * 32];   // pass-2 staging: 64 d x 32 keys (2x4KB)
  __shared__ _Float16 pt[4][16 * LDP];   // per-wave P tiles               (5KB)

  int wv = threadIdx.x >> 5;
  int w = blockIdx.x * 4 + wv;
  int lane = threadIdx.x & 31;
  int nb = w & 127, h = (w >> 7) & 3, b = (w >> 9) & 7;
  int col = lane & 15, hi = lane >> 4;
  int bh = b * NH + h;
  int cidx = wv * 32 + lane;             // 0..127: this thread's 16B-chunk index

  const _Float16* qbase = qh + ((size_t)bh * NN + nb * 16) * DH;
  v16h a0 = load_a_f16(qbase, DH, 0, lane);
  v16h a1 = load_a_f16(qbase, DH, 32, lane);

  float m8[8], l8[8];
#pragma unroll
  for (int g = 0; g < 8; g++) { m8[g] = -1e30f; l8[g] = 0.0f; }

  const _Float16* krbase = krh + (size_t)bh * NN * DH;
  const _Float16* vbase  = vT + (size_t)bh * DH * NN;
  float* sbase = scores + (size_t)bh * NN * NN;

  // ---- pass 1: scores + online stats (kr staged via async->LDS) ----
  async_ld16(lds_off(&kr1[0][cidx * 8]), krbase + (size_t)cidx * 8);
  for (int mb = 0; mb < 128; mb++) {
    wait_async0();
    __syncthreads();                     // everyone's tile mb landed; prior buffer free
    if (mb + 1 < 128)
      async_ld16(lds_off(&kr1[(mb + 1) & 1][cidx * 8]),
                 krbase + (size_t)(mb + 1) * 16 * DH + (size_t)cidx * 8);
    const _Float16* kp = &kr1[mb & 1][0];
    v16h b0 = load_b_f16(kp, DH, 0, lane);
    v16h b1 = load_b_f16(kp, DH, 32, lane);
    v8f cc = {};
    cc = wmma_f16(a0, b0, cc);
    cc = wmma_f16(a1, b1, cc);
#pragma unroll
    for (int g = 0; g < 8; g++) {
      float s = cc[g] * 0.125f;
      __builtin_nontemporal_store(
          s, &sbase[(size_t)(nb * 16 + g + hi * 8) * NN + mb * 16 + col]);
      float mn = fmaxf(m8[g], s);
      l8[g] = l8[g] * __expf(m8[g] - mn) + __expf(s - mn);
      m8[g] = mn;
    }
  }
  // merge (m,l) across the 16 lanes holding each row
#pragma unroll
  for (int g = 0; g < 8; g++) {
#pragma unroll
    for (int mask = 1; mask < 16; mask <<= 1) {
      float om = __shfl_xor(m8[g], mask, 32);
      float ol = __shfl_xor(l8[g], mask, 32);
      float mn = fmaxf(m8[g], om);
      l8[g] = l8[g] * __expf(m8[g] - mn) + ol * __expf(om - mn);
      m8[g] = mn;
    }
  }
  float rl8[8];
#pragma unroll
  for (int g = 0; g < 8; g++) rl8[g] = 1.0f / l8[g];

  // ---- pass 2: recompute S, P = exp(S-m)/l, O += P @ V ----
  _Float16* myl = &pt[wv][0];
  v8f o0 = {}, o1 = {}, o2 = {}, o3 = {};

  // prologue: stage kr (32 rows, contiguous 4KB) + vT (64 rows x 64B) for mb2=0
#pragma unroll
  for (int rr = 0; rr < 2; rr++) {
    int idx = cidx + rr * 128;                        // 0..255
    async_ld16(lds_off(&kr2[0][idx * 8]), krbase + (size_t)idx * 8);
    int vrow = idx >> 2, vsub = idx & 3;
    async_ld16(lds_off(&vt2[0][vrow * 32 + vsub * 8]),
               vbase + (size_t)vrow * NN + vsub * 8);
  }

  for (int mb2 = 0; mb2 < 64; mb2++) {
    wait_async0();
    __syncthreads();
    int cur = mb2 & 1;
    if (mb2 + 1 < 64) {
      int nxt = cur ^ 1;
#pragma unroll
      for (int rr = 0; rr < 2; rr++) {
        int idx = cidx + rr * 128;
        async_ld16(lds_off(&kr2[nxt][idx * 8]),
                   krbase + (size_t)(mb2 + 1) * 32 * DH + (size_t)idx * 8);
        int vrow = idx >> 2, vsub = idx & 3;
        async_ld16(lds_off(&vt2[nxt][vrow * 32 + vsub * 8]),
                   vbase + (size_t)vrow * NN + (mb2 + 1) * 32 + vsub * 8);
      }
    }
    // recompute two 16-col S tiles from staged kr, write P into per-wave LDS
#pragma unroll
    for (int t = 0; t < 2; t++) {
      const _Float16* kp = &kr2[cur][t * 16 * DH];
      v16h b0 = load_b_f16(kp, DH, 0, lane);
      v16h b1 = load_b_f16(kp, DH, 32, lane);
      v8f cc = {};
      cc = wmma_f16(a0, b0, cc);
      cc = wmma_f16(a1, b1, cc);
#pragma unroll
      for (int g = 0; g < 8; g++) {
        float p = __expf(cc[g] * 0.125f - m8[g]) * rl8[g];
        myl[(g + hi * 8) * LDP + t * 16 + col] = (_Float16)p;
      }
    }
    __syncthreads();                    // P visible; staged tiles stable
    v16h ap  = load_a_f16(myl, LDP, 0, lane);
    v16h bv0 = load_b_f16(&vt2[cur][0 * 16 * 32], 32, 0, lane);
    v16h bv1 = load_b_f16(&vt2[cur][1 * 16 * 32], 32, 0, lane);
    v16h bv2 = load_b_f16(&vt2[cur][2 * 16 * 32], 32, 0, lane);
    v16h bv3 = load_b_f16(&vt2[cur][3 * 16 * 32], 32, 0, lane);
    o0 = wmma_f16(ap, bv0, o0);
    o1 = wmma_f16(ap, bv1, o1);
    o2 = wmma_f16(ap, bv2, o2);
    o3 = wmma_f16(ap, bv3, o3);
  }

  // x[b, n, h*64 + d] (torch.cat head order)
  _Float16* xp = xbuf + ((size_t)(b * NN + nb * 16)) * DM + h * DH;
#pragma unroll
  for (int g = 0; g < 8; g++) {
    int row = g + hi * 8;
    xp[(size_t)row * DM + 0  + col] = (_Float16)o0[g];
    xp[(size_t)row * DM + 16 + col] = (_Float16)o1[g];
    xp[(size_t)row * DM + 32 + col] = (_Float16)o2[g];
    xp[(size_t)row * DM + 48 + col] = (_Float16)o3[g];
  }
}

// ---------------------------------------------------------------------------
// Kernel 3: out = x @ Wo^T + bo (contraction dim contiguous in Wo[out,in]).
// ---------------------------------------------------------------------------
__global__ __launch_bounds__(256) void outproj_kernel(const _Float16* xbuf, const float* Wo,
                                                      const float* bo, float* out) {
  int w = blockIdx.x * (blockDim.x >> 5) + (threadIdx.x >> 5);
  int lane = threadIdx.x & 31;
  int ot = w & 15;
  int rb = w >> 4;
  int col = lane & 15, hi = lane >> 4;

  const _Float16* A = xbuf + (size_t)rb * 16 * DM;
  const float* Bw = Wo + (size_t)(ot * 16) * DM;
  v8f c = {};
#pragma unroll
  for (int kc = 0; kc < 8; kc++)
    c = wmma_f16(load_a_f16(A, DM, kc * 32, lane), load_b_f32(Bw, DM, kc * 32, lane), c);

  float bias = bo[ot * 16 + col];
  float* op = out + (size_t)rb * 16 * DM + ot * 16;
#pragma unroll
  for (int g = 0; g < 8; g++)
    __builtin_nontemporal_store(c[g] + bias, &op[(size_t)(g + hi * 8) * DM + col]);
}

// ---------------------------------------------------------------------------
extern "C" void kernel_launch(void* const* d_in, const int* in_sizes, int n_in,
                              void* d_out, int out_size, void* d_ws, size_t ws_size,
                              hipStream_t stream) {
  (void)in_sizes; (void)n_in; (void)out_size; (void)ws_size;
  const float* Q  = (const float*)d_in[0];
  const float* K  = (const float*)d_in[1];
  const float* V  = (const float*)d_in[2];
  const float* R  = (const float*)d_in[3];
  const float* Wq = (const float*)d_in[4];
  const float* bq = (const float*)d_in[5];
  const float* Wk = (const float*)d_in[6];
  const float* bk = (const float*)d_in[7];
  const float* Wv = (const float*)d_in[8];
  const float* bv = (const float*)d_in[9];
  const float* Wr = (const float*)d_in[10];
  const float* br = (const float*)d_in[11];
  const float* Wo = (const float*)d_in[12];
  const float* bo = (const float*)d_in[13];

  float* out    = (float*)d_out;
  float* scores = out + (size_t)BB * NN * DM;

  char* wsb = (char*)d_ws;
  _Float16* WtT = (_Float16*)wsb;
  _Float16* qhp = (_Float16*)(wsb + (1 << 19));
  const size_t perT = (size_t)BB * NH * NN * DH;  // 4,194,304 halves
  _Float16* krp = qhp + perT;
  _Float16* vTp = krp + perT;
  _Float16* xbp = vTp + perT;

  prep_weights<<<1024, 256, 0, stream>>>(Wq, Wk, Wv, Wr, WtT);
  proj_kernel<<<6144, 256, 0, stream>>>(Q, K, V, R, bq, bk, bv, br, WtT, qhp, krp, vTp);
  attn_kernel<<<1024, 128, 0, stream>>>(qhp, krp, vTp, scores, xbp);
  outproj_kernel<<<2048, 256, 0, stream>>>(xbp, Wo, bo, out);
}
